// GridSampleVT_76768245449524
// MI455X (gfx1250) — matrix operand order repaired and checked by reading, not compiled, not run on hardware
//
#include <hip/hip_runtime.h>
#include <hip/hip_bf16.h>

// ---- problem constants (from reference) ----
#define NCAM   4
#define CCH    64
#define IMH    28
#define IMW    60
#define ZCD    200
#define YCD    8
#define XCD    200
#define OUTC   128
#define KDIM   (CCH * YCD)        // 512
#define ZX     (ZCD * XCD)        // 40000
#define NVOX   (ZX * YCD)         // 320000
#define CHW    (IMH * IMW)        // 1680
#define EPSV   1e-6f

typedef __attribute__((ext_vector_type(16))) _Float16 v16h;
typedef __attribute__((ext_vector_type(8)))  _Float16 v8h;
typedef __attribute__((ext_vector_type(8)))  float    v8f;

// K-permutation shared by A and B operands:  k' = y*64 + c   (k_orig = c*8 + y)
// The GEMM result is invariant to any permutation applied to both operands;
// this one makes each sampler thread's 64 channel outputs contiguous in T.

// ---------------------------------------------------------------------------
// Kernel 1: pack W_comp (128x512 f32, k_orig = c*8+y) into f16 WMMA-A
// fragments under the permuted k' ordering.
//   Wsw[((ot*16 + kt)*32 + lane)*16 + j]
//   A 16x32 f16 layout as used here (contiguous k, not ISA order): lane M = lane%16,
//   k'(j) = kt*32 + (lane/16)*16 + j
// ---------------------------------------------------------------------------
__global__ void __launch_bounds__(256)
pack_w_kernel(const float* __restrict__ Wc, _Float16* __restrict__ Wsw) {
    int t = blockIdx.x * blockDim.x + threadIdx.x;   // 8*16*32 = 4096
    if (t >= (OUTC / 16) * (KDIM / 32) * 32) return;
    int lane = t & 31;
    int kt   = (t >> 5) & 15;
    int ot   = t >> 9;
    int m    = ot * 16 + (lane & 15);
    _Float16* dst = Wsw + (long)t * 16;
#pragma unroll
    for (int j = 0; j < 16; ++j) {
        int kp = kt * 32 + ((lane >> 4) << 4) + j; // k'
        int c  = kp & (CCH - 1);
        int yy = kp >> 6;
        dst[j] = (_Float16)Wc[m * KDIM + c * YCD + yy];                  // k_orig
    }
}

// ---------------------------------------------------------------------------
// Kernel 2: transpose img_feats (N,C,H,W) f32 -> channel-last f16 [n][h][w][c]
// so each bilinear corner's 64 channels are one contiguous 128 B run.
// ---------------------------------------------------------------------------
__global__ void __launch_bounds__(256)
feats_chlast_kernel(const float* __restrict__ f, _Float16* __restrict__ o) {
    int t = blockIdx.x * blockDim.x + threadIdx.x;   // NCAM*CHW = 6720
    if (t >= NCAM * CHW) return;
    int n  = t / CHW;
    int hw = t - n * CHW;
    const float* src = f + (long)n * CCH * CHW + hw;
    _Float16*    dst = o + (long)t * CCH;
#pragma unroll 8
    for (int c = 0; c < CCH; ++c) dst[c] = (_Float16)src[c * CHW];
}

// ---------------------------------------------------------------------------
// Kernel 3: grid-sample (D=1 trilinear) + masked camera mean.
// One thread per (zx, y): 16 corner slots (4 cams x 4 in-plane corners,
// z/bounds validity folded into the weight), 8x v8h gathers per slot,
// f32 accumulation, 8 coalesced v8h stores to T[zx][y*64 .. y*64+63].
// ---------------------------------------------------------------------------
__global__ void __launch_bounds__(256)
sample_reduce_kernel(const float* __restrict__ coords,
                     const float* __restrict__ valid,
                     const _Float16* __restrict__ FH,   // channel-last f16 feats
                     _Float16* __restrict__ T) {
    int tid = blockIdx.x * blockDim.x + threadIdx.x;
    if (tid >= NVOX) return;
    int y  = tid & (YCD - 1);
    int zx = tid >> 3;
    int z  = zx / XCD;
    int x  = zx - z * XCD;

    long vox = ((long)z * YCD + y) * XCD + x;        // index within (ZC,YC,XC)

    float wgt[NCAM * 4];
    int   off[NCAM * 4];
    float den = 0.f;

#pragma unroll
    for (int n = 0; n < NCAM; ++n) {
        long cb = (long)n * (ZCD * YCD * XCD) + vox;
        float gx = coords[cb * 3 + 0];
        float gy = coords[cb * 3 + 1];
        float gz = coords[cb * 3 + 2];
        float m  = valid[cb];                        // exactly 0.0 or 1.0
        den += m;

        float fx = ((gx + 1.f) * (float)IMW - 1.f) * 0.5f;
        float fy = ((gy + 1.f) * (float)IMH - 1.f) * 0.5f;
        float fz = ((gz + 1.f) * 1.f        - 1.f) * 0.5f;   // D == 1
        float x0f = floorf(fx), y0f = floorf(fy), z0f = floorf(fz);
        float wx = fx - x0f, wy = fy - y0f, wz = fz - z0f;
        int x0 = (int)x0f, y0 = (int)y0f, z0 = (int)z0f;

        // Only depth slice 0 exists: z0==0 -> (1-wz); z0==-1 -> wz.
        float zw = (z0 == 0) ? (1.f - wz) : ((z0 == -1) ? wz : 0.f);
        float wm = m * zw;                           // mask^2 == mask

#pragma unroll
        for (int j = 0; j < 4; ++j) {
            int dy = j >> 1, dx = j & 1;
            int yi = y0 + dy, xi = x0 + dx;
            bool ok = (yi >= 0) & (yi < IMH) & (xi >= 0) & (xi < IMW);
            float w = ok ? wm * (dy ? wy : 1.f - wy) * (dx ? wx : 1.f - wx) : 0.f;
            wgt[n * 4 + j] = w;
            off[n * 4 + j] = ok ? ((n * IMH + yi) * IMW + xi) * CCH : 0;
        }
    }

    float acc[CCH];
#pragma unroll
    for (int c = 0; c < CCH; ++c) acc[c] = 0.f;

#pragma unroll
    for (int i = 0; i < NCAM * 4; ++i) {
        float w = wgt[i];
        const v8h* p = (const v8h*)(FH + off[i]);
#pragma unroll
        for (int g = 0; g < 8; ++g) {
            v8h h = p[g];
#pragma unroll
            for (int e = 0; e < 8; ++e)
                acc[g * 8 + e] = fmaf((float)h[e], w, acc[g * 8 + e]); // fma_mix
        }
    }

    float inv = 1.f / (den + EPSV);
    _Float16* dst = T + (long)zx * KDIM + y * CCH;   // k' = y*64 + c
#pragma unroll
    for (int g = 0; g < 8; ++g) {
        v8h o;
#pragma unroll
        for (int e = 0; e < 8; ++e) o[e] = (_Float16)(acc[g * 8 + e] * inv);
        *(v8h*)(dst + g * 8) = o;
    }
}

// ---------------------------------------------------------------------------
// Kernel 4: out[o, zx] = sum_k' A[o,k'] * T[zx,k'] + bias[o]
// M=128, K=512, N=40000.  Block = 8 waves; each wave owns a DISTINCT 16-wide
// zx tile and computes all 8 o-tiles (8 accumulators) -> the 41 MB B stream
// is loaded exactly once; only the tiny L2-resident A (128 KB) is re-read.
// 8x v_wmma_f32_16x16x32_f16 per K-step, 16 K-steps.
// B layout (same contiguous k as A, not ISA order): lane N = l%16, k = (l>=16)*16 + 2v + h -> one contiguous
// v16h per fragment at T[(zx0+l%16)*512 + kt*32 + (l/16)*16].
// ---------------------------------------------------------------------------
__global__ void __launch_bounds__(256)
wmma_gemm_kernel(const _Float16* __restrict__ Wsw,
                 const _Float16* __restrict__ T,
                 const float*    __restrict__ bias,
                 float*          __restrict__ out) {
    int lane = threadIdx.x & 31;
    int wave = threadIdx.x >> 5;
    int zxt  = blockIdx.x * 8 + wave;                // zx tile per wave
    if (zxt >= ZX / 16) return;
    int zx0 = zxt * 16;
    int ln  = lane & 15;
    int lh  = lane >> 4;

    v8f acc[8];
#pragma unroll
    for (int i = 0; i < 8; ++i) acc[i] = v8f{0.f, 0.f, 0.f, 0.f, 0.f, 0.f, 0.f, 0.f};

    const _Float16* brow  = T + (long)(zx0 + ln) * KDIM + lh * 16;
    const _Float16* abase = Wsw + (long)lane * 16;

    for (int kt = 0; kt < KDIM / 32; ++kt) {
        v16h b = *(const v16h*)(brow + kt * 32);
        const _Float16* ak = abase + kt * (32 * 16);
#pragma unroll
        for (int ot = 0; ot < 8; ++ot) {
            v16h a = *(const v16h*)(ak + ot * (16 * 32 * 16));
            acc[ot] = __builtin_amdgcn_wmma_f32_16x16x32_f16(
                /*neg_a=*/false, a, /*neg_b=*/false, b,
                /*c_mod=*/(short)0, acc[ot], /*reuse_a=*/false, /*reuse_b=*/false);
        }
    }

    // C/D layout: VGPR r holds M = r + 8*(lane/16), N = lane%16.
#pragma unroll
    for (int ot = 0; ot < 8; ++ot) {
        int o0 = ot * 16 + lh * 8;
#pragma unroll
        for (int r = 0; r < 8; ++r) {
            int o = o0 + r;
            out[(long)o * ZX + zx0 + ln] = acc[ot][r] + bias[o];
        }
    }
}

// ---------------------------------------------------------------------------
extern "C" void kernel_launch(void* const* d_in, const int* in_sizes, int n_in,
                              void* d_out, int out_size, void* d_ws, size_t ws_size,
                              hipStream_t stream) {
    const float* coords = (const float*)d_in[0];   // (1,4,200,8,200,3)
    const float* validm = (const float*)d_in[1];   // (1,4,200,8,200,1)
    const float* feats  = (const float*)d_in[2];   // (1,4,64,28,60)
    const float* Wc     = (const float*)d_in[3];   // (128,512)
    const float* bias   = (const float*)d_in[4];   // (128,)
    float* out = (float*)d_out;                    // (1,1,128,200,200)

    const size_t WSW_BYTES = (size_t)OUTC * KDIM * 2;          // 131072
    const size_t FH_BYTES  = (size_t)NCAM * CHW * CCH * 2;     // 860160
    _Float16* Wsw = (_Float16*)d_ws;
    _Float16* FH  = (_Float16*)((char*)d_ws + WSW_BYTES);
    _Float16* T   = (_Float16*)((char*)d_ws + WSW_BYTES + FH_BYTES);

    // 1) pack W into permuted-K WMMA-A fragment order (f16)
    pack_w_kernel<<<16, 256, 0, stream>>>(Wc, Wsw);

    // 2) features -> channel-last f16
    feats_chlast_kernel<<<(NCAM * CHW + 255) / 256, 256, 0, stream>>>(feats, FH);

    // 3) grid-sample + camera reduction -> T[zx][k'] f16
    sample_reduce_kernel<<<NVOX / 256, 256, 0, stream>>>(coords, validm, FH, T);

    // 4) WMMA GEMM: out = W @ T^T + b
    wmma_gemm_kernel<<<(ZX / 16 + 7) / 8, 256, 0, stream>>>(Wsw, T, bias, out);
}
